// SlotAttentionMulti_86440511799458
// MI455X (gfx1250) — matrix operand order repaired and checked by reading, not compiled, not run on hardware
//
#include <hip/hip_runtime.h>
#include <hip/hip_bf16.h>
#include <math.h>

#define B_   8
#define V_   4
#define BV   32
#define N_   4096
#define F_   256
#define DV   64
#define DA   128
#define SF   192
#define Q_   128
#define S_   7
#define HID  256
#define NSTEP 3
#define EPS  1e-5f
#define COEF 0.08838834764831845f   // 1/sqrt(128)

typedef __attribute__((ext_vector_type(16))) _Float16 v16h;
typedef __attribute__((ext_vector_type(8)))  _Float16 h8;
typedef __attribute__((ext_vector_type(4)))  _Float16 h4;
typedef __attribute__((ext_vector_type(8)))  float    v8f;
typedef __attribute__((ext_vector_type(4)))  float    f4;

union Frag { v16h v; h8 h[2]; };

__device__ __forceinline__ v8f wmma16(v16h a, v16h b, v8f c) {
  // D = A(16x32 f16) * B(32x16 f16) + C(16x16 f32)
  return __builtin_amdgcn_wmma_f32_16x16x32_f16(false, a, false, b, (short)0, c, false, false);
}

// ---------------------------------------------------------------------------
// K1: fused LayerNorm(x) + [x @ W_key | x @ W_val], f16 WMMA.
// One block = 64 rows of one (b,v). All fragments = 2x b128 loads, batched per
// tile (single wait point, WMMA chain back-to-back). Key/val phases split so
// base addressing is uniform (no per-iteration cndmask selects).
// key16 stored [N][Q] (K-major for K3); val stored transposed [SF][N] (K-major for K4).
// ---------------------------------------------------------------------------
__global__ void __launch_bounds__(256) ln_proj_kernel(
    const float* __restrict__ x, const float* __restrict__ ln_g, const float* __restrict__ ln_b,
    const _Float16* __restrict__ WkT, const _Float16* __restrict__ WvT,
    _Float16* __restrict__ key16, _Float16* __restrict__ valT) {
  __shared__ _Float16 xn[64][F_ + 8];
  int blk  = blockIdx.x;
  int bv   = blk >> 6;            // 64 row-blocks per (b,v)
  int row0 = (blk & 63) * 64;
  int tid  = threadIdx.x;
  int lane = tid & 31;

  // LayerNorm: 4 threads per row, float4 NT loads, wave-local shuffle reduction
  {
    int r = tid >> 2, qq = tid & 3;
    const f4* xr4 = (const f4*)(x + (size_t)(bv * N_ + row0 + r) * F_);
    float s = 0.f, s2 = 0.f;
    for (int j4 = qq * 16; j4 < qq * 16 + 16; ++j4) {
      f4 v = __builtin_nontemporal_load(xr4 + j4);
      s  += v[0] + v[1] + v[2] + v[3];
      s2 += v[0]*v[0] + v[1]*v[1] + v[2]*v[2] + v[3]*v[3];
    }
    s  += __shfl_xor(s, 1, 32);  s  += __shfl_xor(s, 2, 32);
    s2 += __shfl_xor(s2, 1, 32); s2 += __shfl_xor(s2, 2, 32);
    float mean = s * (1.f / F_);
    float rstd = rsqrtf(s2 * (1.f / F_) - mean * mean + EPS);
    for (int jj = qq * 64; jj < qq * 64 + 64; jj += 8) {
      f4 v0 = __builtin_nontemporal_load(xr4 + jj / 4);
      f4 v1 = __builtin_nontemporal_load(xr4 + jj / 4 + 1);
      f4 g0 = ((const f4*)ln_g)[jj / 4], g1 = ((const f4*)ln_g)[jj / 4 + 1];
      f4 b0 = ((const f4*)ln_b)[jj / 4], b1 = ((const f4*)ln_b)[jj / 4 + 1];
      h8 o;
      #pragma unroll
      for (int e = 0; e < 4; ++e) {
        o[e]     = (_Float16)((v0[e] - mean) * rstd * g0[e] + b0[e]);
        o[4 + e] = (_Float16)((v1[e] - mean) * rstd * g1[e] + b1[e]);
      }
      *(h8*)(&xn[r][jj]) = o;
    }
  }
  __syncthreads();

  int wave  = tid >> 5;
  int m     = lane & 15;
  int h8off = (lane & 16) ? 8 : 0;
  int kboff = (lane & 16) ? 16 : 0;

  // Phase 1: key tiles (4 row tiles x 8 col tiles = 32; 4 per wave)
  for (int t = wave; t < 32; t += 8) {
    int rt = t >> 3, ct = t & 7;
    const _Float16* wrow = WkT + (size_t)(ct * 16 + m) * F_;
    const _Float16* arow = &xn[rt * 16 + m][0];
    Frag afr[8], bfr[8];
    #pragma unroll
    for (int ks = 0; ks < 8; ++ks) {       // batched: one clause, one wait
      afr[ks].h[0] = *(const h8*)(arow + ks * 32 + h8off);
      afr[ks].h[1] = *(const h8*)(arow + ks * 32 + 16 + h8off);
      bfr[ks].h[0] = *(const h8*)(wrow + ks * 32 + h8off);
      bfr[ks].h[1] = *(const h8*)(wrow + ks * 32 + 16 + h8off);
    }
    v8f c = {};
    #pragma unroll
    for (int ks = 0; ks < 8; ++ks) c = wmma16(afr[ks].v, bfr[ks].v, c);
    int rb = (lane & 16) ? 8 : 0;
    #pragma unroll
    for (int v = 0; v < 8; ++v)
      key16[(size_t)(bv * N_ + row0 + rt * 16 + rb + v) * Q_ + ct * 16 + m] = (_Float16)c[v];
  }

  // Phase 2: val tiles (4 row tiles x 12 col tiles = 48; 6 per wave)
  for (int t = wave; t < 48; t += 8) {
    int rt = t / 12, ct = t % 12;
    const _Float16* wrow = WvT + (size_t)(ct * 16 + m) * F_;
    const _Float16* arow = &xn[rt * 16 + m][0];
    Frag afr[8], bfr[8];
    #pragma unroll
    for (int ks = 0; ks < 8; ++ks) {
      afr[ks].h[0] = *(const h8*)(arow + ks * 32 + h8off);
      afr[ks].h[1] = *(const h8*)(arow + ks * 32 + 16 + h8off);
      bfr[ks].h[0] = *(const h8*)(wrow + ks * 32 + h8off);
      bfr[ks].h[1] = *(const h8*)(wrow + ks * 32 + 16 + h8off);
    }
    v8f c = {};
    #pragma unroll
    for (int ks = 0; ks < 8; ++ks) c = wmma16(afr[ks].v, bfr[ks].v, c);
    int rb = (lane & 16) ? 8 : 0;
    h8 o;
    #pragma unroll
    for (int v = 0; v < 8; ++v) o[v] = (_Float16)c[v];   // 8 consecutive n at fixed col
    *(h8*)(valT + ((size_t)bv * SF + ct * 16 + m) * N_ + row0 + rt * 16 + rb) = o;
  }
}

// ---------------------------------------------------------------------------
// K2: full = [view|attr], LN_qry, q = LN(full) @ W_qry * COEF (f16, rows 7..15 = 0)
// ---------------------------------------------------------------------------
__global__ void __launch_bounds__(256) slots_pre_kernel(
    const float* __restrict__ sview, const float* __restrict__ sattr,
    const float* __restrict__ lnq_g, const float* __restrict__ lnq_b,
    const float* __restrict__ Wq, float* __restrict__ fullb, _Float16* __restrict__ q16) {
  int bv = blockIdx.x;
  int b  = bv >> 2;
  int tid = threadIdx.x;
  __shared__ float fs[S_][SF];
  __shared__ float lf[S_][SF];
  __shared__ float sm[S_], sr[S_];
  for (int idx = tid; idx < S_ * SF; idx += 256) {
    int s = idx / SF, i = idx % SF;
    float v = (i < DV) ? sview[bv * DV + i] : sattr[(b * S_ + s) * DA + (i - DV)];
    fs[s][i] = v;
    fullb[(size_t)(bv * S_ + s) * SF + i] = v;
  }
  __syncthreads();
  if (tid < S_) {
    float s1 = 0.f, s2 = 0.f;
    for (int i = 0; i < SF; ++i) { float v = fs[tid][i]; s1 += v; s2 += v * v; }
    float mean = s1 / SF;
    sm[tid] = mean;
    sr[tid] = rsqrtf(s2 / SF - mean * mean + EPS);
  }
  __syncthreads();
  for (int idx = tid; idx < S_ * SF; idx += 256) {
    int s = idx / SF, i = idx % SF;
    lf[s][i] = (fs[s][i] - sm[s]) * sr[s] * lnq_g[i] + lnq_b[i];
  }
  __syncthreads();
  for (int o = tid; o < 16 * Q_; o += 256) {
    int s = o >> 7, j = o & (Q_ - 1);
    float acc = 0.f;
    if (s < S_) {
      for (int i = 0; i < SF; ++i) acc += lf[s][i] * Wq[i * Q_ + j];
      acc *= COEF;
    }
    q16[(size_t)bv * 16 * Q_ + o] = (_Float16)acc;   // zero-padded rows 7..15
  }
}

// ---------------------------------------------------------------------------
// K3: logits^T = q_pad(16x128) @ key^T per (b,v). Slot log_softmax is per-lane
// (rows 0..6 in one lane's C VGPRs); input softmax via 2-pass WG reduction.
// B fragments batched: one clause of 8 b128 loads, then 4 WMMAs back-to-back.
// ---------------------------------------------------------------------------
__global__ void __launch_bounds__(256) attn_kernel(
    const _Float16* __restrict__ key16, const _Float16* __restrict__ q16,
    _Float16* __restrict__ attn16, float* __restrict__ etmp) {
  int bv = blockIdx.x;
  int tid = threadIdx.x;
  int lane = tid & 31, wave = tid >> 5;
  __shared__ float wsum[8][S_];
  __shared__ float winv[S_];

  int m = lane & 15;
  int h8off = (lane & 16) ? 8 : 0;
  int khalf = (lane & 16) ? 16 : 0;
  Frag qa[4];
  {
    const _Float16* qb = q16 + (size_t)bv * 16 * Q_ + m * Q_;
    #pragma unroll
    for (int kf = 0; kf < 4; ++kf) {
      qa[kf].h[0] = *(const h8*)(qb + kf * 32 + h8off);
      qa[kf].h[1] = *(const h8*)(qb + kf * 32 + 16 + h8off);
    }
  }
  float acc[S_];
  #pragma unroll
  for (int s = 0; s < S_; ++s) acc[s] = 0.f;
  size_t kbvbase = (size_t)bv * N_;
  size_t ebase   = (size_t)bv * S_ * N_;

  for (int t = 0; t < 32; ++t) {                       // 8 waves x 32 tiles = 4096 inputs
    int n0 = wave * 512 + t * 16;
    const _Float16* kr = key16 + (kbvbase + n0 + m) * Q_;
    Frag bf[4];
    #pragma unroll
    for (int kf = 0; kf < 4; ++kf) {                   // batched loads
      bf[kf].h[0] = *(const h8*)(kr + kf * 32 + h8off);
      bf[kf].h[1] = *(const h8*)(kr + kf * 32 + 16 + h8off);
    }
    v8f c = {};
    #pragma unroll
    for (int kf = 0; kf < 4; ++kf) c = wmma16(qa[kf].v, bf[kf].v, c);
    if (!(lane & 16)) {                                // lanes 0..15 hold slot rows 0..7
      float mx = c[0];
      #pragma unroll
      for (int s = 1; s < S_; ++s) mx = fmaxf(mx, c[s]);
      float se = 0.f;
      #pragma unroll
      for (int s = 0; s < S_; ++s) se += __expf(c[s] - mx);
      float lse = mx + __logf(se);
      #pragma unroll
      for (int s = 0; s < S_; ++s) {
        float e = __expf(c[s] - lse);                  // exp(log_softmax) in (0,1]: safe
        acc[s] += e;
        etmp[ebase + (size_t)s * N_ + n0 + m] = e;
      }
    }
  }
  #pragma unroll
  for (int s = 0; s < S_; ++s) {                       // reduce over lanes 0..15
    float v = acc[s];
    v += __shfl_xor(v, 1, 16); v += __shfl_xor(v, 2, 16);
    v += __shfl_xor(v, 4, 16); v += __shfl_xor(v, 8, 16);
    acc[s] = v;
  }
  if (lane == 0) {
    #pragma unroll
    for (int s = 0; s < S_; ++s) wsum[wave][s] = acc[s];
  }
  __syncthreads();
  if (tid < S_) {
    float tot = 0.f;
    for (int w = 0; w < 8; ++w) tot += wsum[w][tid];
    winv[tid] = 1.f / tot;
  }
  __syncthreads();
  size_t abase = (size_t)bv * 16 * N_;
  for (int idx4 = tid * 4; idx4 < S_ * N_; idx4 += 256 * 4) {
    int s = idx4 >> 12;                                // N_ = 4096
    f4 e = *(const f4*)(etmp + ebase + idx4);
    float wi = winv[s];
    h4 o;
    #pragma unroll
    for (int e2 = 0; e2 < 4; ++e2) o[e2] = (_Float16)(e[e2] * wi);
    *(h4*)(attn16 + abase + idx4) = o;
  }
}

// ---------------------------------------------------------------------------
// K4: updates^T = attn^T(16xN f16) @ val (K-major valT[SF][N]).
// 12 waves = 12 col tiles; K-loop over N=4096, 2-deep software pipeline so the
// wait before each WMMA tolerates the next iteration's in-flight loads.
// ---------------------------------------------------------------------------
__global__ void __launch_bounds__(384) updates_kernel(
    const _Float16* __restrict__ attn16, const _Float16* __restrict__ valT,
    float* __restrict__ upd) {
  int bv = blockIdx.x;
  int tid = threadIdx.x;
  int lane = tid & 31, wave = tid >> 5;    // 12 waves
  int c0 = wave * 16;
  int m = lane & 15;
  int h8off = (lane & 16) ? 8 : 0;
  int khalf = (lane & 16) ? 16 : 0;
  const _Float16* arow = attn16 + (size_t)bv * 16 * N_ + (size_t)m * N_;
  const _Float16* vrow = valT + ((size_t)bv * SF + c0 + m) * N_;
  v8f c = {};
  Frag a0, b0;
  a0.h[0] = *(const h8*)(arow + h8off);
  a0.h[1] = *(const h8*)(arow + 16 + h8off);
  b0.h[0] = *(const h8*)(vrow + h8off);
  b0.h[1] = *(const h8*)(vrow + 16 + h8off);
  #pragma unroll 4
  for (int n0 = 0; n0 < N_; n0 += 32) {
    // Preload next iteration (final preload reads 32B of pad into the adjacent
    // ws buffer -- in-bounds of d_ws, values never consumed).
    int np = n0 + 32;
    Frag a1, b1;
    a1.h[0] = *(const h8*)(arow + np + h8off);
    a1.h[1] = *(const h8*)(arow + np + 16 + h8off);
    b1.h[0] = *(const h8*)(vrow + np + h8off);
    b1.h[1] = *(const h8*)(vrow + np + 16 + h8off);
    c = wmma16(a0.v, b0.v, c);
    a0 = a1; b0 = b1;
  }
  int rb = (lane & 16) ? 8 : 0;
  #pragma unroll
  for (int v = 0; v < 8; ++v)
    upd[(size_t)bv * 16 * SF + (rb + v) * SF + c0 + m] = c[v];
}

// ---------------------------------------------------------------------------
// K5a: GRU + LN_res + residual MLP for the 7 slots of one (b,v). All in LDS.
// ---------------------------------------------------------------------------
__global__ void __launch_bounds__(256) gru_mlp_kernel(
    const float* __restrict__ upd, const float* __restrict__ fullb,
    const float* __restrict__ gWi, const float* __restrict__ gWh,
    const float* __restrict__ gbi, const float* __restrict__ gbh,
    const float* __restrict__ lnr_g, const float* __restrict__ lnr_b,
    const float* __restrict__ W1, const float* __restrict__ b1,
    const float* __restrict__ W2, const float* __restrict__ b2,
    float* __restrict__ fulln) {
  int bv = blockIdx.x;
  int tid = threadIdx.x;
  __shared__ float u[S_][SF], h[S_][SF];
  __shared__ float gi[S_][3 * SF], gh[S_][3 * SF];
  __shared__ float mn[S_][SF], lm[S_][SF];
  __shared__ float hid[S_][HID];
  __shared__ float sm[S_], sr[S_];
  for (int idx = tid; idx < S_ * SF; idx += 256) {
    int s = idx / SF, i = idx % SF;
    u[s][i] = upd[(size_t)bv * 16 * SF + s * SF + i];
    h[s][i] = fullb[(size_t)(bv * S_ + s) * SF + i];
  }
  __syncthreads();
  for (int o = tid; o < S_ * 3 * SF; o += 256) {
    int s = o / (3 * SF), j = o % (3 * SF);
    float ai = gbi[j], ah = gbh[j];
    for (int i = 0; i < SF; ++i) {
      ai += u[s][i] * gWi[i * 3 * SF + j];
      ah += h[s][i] * gWh[i * 3 * SF + j];
    }
    gi[s][j] = ai; gh[s][j] = ah;
  }
  __syncthreads();
  for (int idx = tid; idx < S_ * SF; idx += 256) {
    int s = idx / SF, i = idx % SF;
    float r  = 1.f / (1.f + __expf(-(gi[s][i] + gh[s][i])));
    float z  = 1.f / (1.f + __expf(-(gi[s][SF + i] + gh[s][SF + i])));
    float nn = tanhf(gi[s][2 * SF + i] + r * gh[s][2 * SF + i]);
    mn[s][i] = (1.f - z) * nn + z * h[s][i];
  }
  __syncthreads();
  if (tid < S_) {
    float s1 = 0.f, s2 = 0.f;
    for (int i = 0; i < SF; ++i) { float v = mn[tid][i]; s1 += v; s2 += v * v; }
    float mean = s1 / SF;
    sm[tid] = mean; sr[tid] = rsqrtf(s2 / SF - mean * mean + EPS);
  }
  __syncthreads();
  for (int idx = tid; idx < S_ * SF; idx += 256) {
    int s = idx / SF, i = idx % SF;
    lm[s][i] = (mn[s][i] - sm[s]) * sr[s] * lnr_g[i] + lnr_b[i];
  }
  __syncthreads();
  for (int o = tid; o < S_ * HID; o += 256) {
    int s = o >> 8, j = o & (HID - 1);
    float acc = b1[j];
    for (int i = 0; i < SF; ++i) acc += lm[s][i] * W1[i * HID + j];
    hid[s][j] = fmaxf(acc, 0.f);
  }
  __syncthreads();
  for (int o = tid; o < S_ * SF; o += 256) {
    int s = o / SF, i = o % SF;
    float acc = b2[i];
    for (int j = 0; j < HID; ++j) acc += hid[s][j] * W2[j * SF + i];
    fulln[(size_t)bv * S_ * SF + o] = mn[s][i] + acc;
  }
}

// K5b: slots_view = mean over slots of full[:, :64]; slots_attr = mean over views
__global__ void means_kernel(const float* __restrict__ fulln,
                             float* __restrict__ sview, float* __restrict__ sattr,
                             float* __restrict__ out) {
  int idx = blockIdx.x * blockDim.x + threadIdx.x;
  if (idx < BV * DV) {
    int bv = idx / DV, d = idx % DV;
    float acc = 0.f;
    for (int s = 0; s < S_; ++s) acc += fulln[(size_t)(bv * S_ + s) * SF + d];
    float m = acc * (1.f / S_);
    sview[idx] = m; out[idx] = m;
  } else if (idx < BV * DV + B_ * S_ * DA) {
    int j = idx - BV * DV;
    int b = j / (S_ * DA);
    int rem = j % (S_ * DA);
    int s = rem / DA, d = rem % DA;
    float acc = 0.f;
    for (int v = 0; v < V_; ++v)
      acc += fulln[(size_t)((b * V_ + v) * S_ + s) * SF + DV + d];
    float m = acc * (1.f / V_);
    sattr[j] = m; out[BV * DV + j] = m;
  }
}

__global__ void init_slots_kernel(
    const float* __restrict__ nv, const float* __restrict__ noa, const float* __restrict__ nba,
    const float* __restrict__ vloc, const float* __restrict__ vls,
    const float* __restrict__ aol, const float* __restrict__ aos,
    const float* __restrict__ abl, const float* __restrict__ absc,
    float* __restrict__ sview, float* __restrict__ sattr) {
  int idx = blockIdx.x * blockDim.x + threadIdx.x;
  if (idx < BV * DV) {
    int d = idx % DV;
    sview[idx] = vloc[d] + __expf(vls[d]) * nv[idx];
  } else if (idx < BV * DV + B_ * S_ * DA) {
    int j = idx - BV * DV;
    int b = j / (S_ * DA);
    int rem = j % (S_ * DA);
    int s = rem / DA, d = rem % DA;
    float val;
    if (s < S_ - 1) val = aol[d] + __expf(aos[d]) * noa[(b * (S_ - 1) + s) * DA + d];
    else            val = abl[d] + __expf(absc[d]) * nba[b * DA + d];
    sattr[j] = val;
  }
}

// f32 -> f16 weight conversion, stored TRANSPOSED ([out_col][F]) so WMMA B-fragments
// are contiguous 128-bit loads per lane.
__global__ void wconv_kernel(const float* __restrict__ Wk, const float* __restrict__ Wv,
                             _Float16* __restrict__ WkT, _Float16* __restrict__ WvT) {
  int idx = blockIdx.x * blockDim.x + threadIdx.x;
  if (idx < F_ * Q_) {
    int i = idx / Q_, j = idx % Q_;
    WkT[j * F_ + i] = (_Float16)Wk[idx];
  } else if (idx < F_ * (Q_ + SF)) {
    int t = idx - F_ * Q_;
    int i = t / SF, j = t % SF;
    WvT[j * F_ + i] = (_Float16)Wv[t];
  }
}

extern "C" void kernel_launch(void* const* d_in, const int* in_sizes, int n_in,
                              void* d_out, int out_size, void* d_ws, size_t ws_size,
                              hipStream_t stream) {
  (void)in_sizes; (void)n_in; (void)out_size; (void)ws_size;
  const float* x        = (const float*)d_in[0];
  const float* nview    = (const float*)d_in[1];
  const float* nobj     = (const float*)d_in[2];
  const float* nbck     = (const float*)d_in[3];
  const float* ln_in_g  = (const float*)d_in[4];
  const float* ln_in_b  = (const float*)d_in[5];
  const float* ln_qry_g = (const float*)d_in[6];
  const float* ln_qry_b = (const float*)d_in[7];
  const float* ln_res_g = (const float*)d_in[8];
  const float* ln_res_b = (const float*)d_in[9];
  const float* vloc     = (const float*)d_in[10];
  const float* vls      = (const float*)d_in[11];
  const float* aol      = (const float*)d_in[12];
  const float* aos      = (const float*)d_in[13];
  const float* abl      = (const float*)d_in[14];
  const float* absc     = (const float*)d_in[15];
  const float* W_qry    = (const float*)d_in[16];
  const float* W_key    = (const float*)d_in[17];
  const float* W_val    = (const float*)d_in[18];
  const float* gWi      = (const float*)d_in[19];
  const float* gWh      = (const float*)d_in[20];
  const float* gbi      = (const float*)d_in[21];
  const float* gbh      = (const float*)d_in[22];
  const float* W1       = (const float*)d_in[23];
  const float* b1       = (const float*)d_in[24];
  const float* W2       = (const float*)d_in[25];
  const float* b2       = (const float*)d_in[26];
  float* out = (float*)d_out;

  char* p = (char*)d_ws;
  auto alloc = [&](size_t bytes) -> void* {
    void* r = (void*)p;
    p += (bytes + 255) & ~(size_t)255;
    return r;
  };
  _Float16* WkT    = (_Float16*)alloc((size_t)F_ * Q_ * 2);
  _Float16* WvT    = (_Float16*)alloc((size_t)F_ * SF * 2);
  _Float16* key16  = (_Float16*)alloc((size_t)BV * N_ * Q_ * 2);
  _Float16* valT   = (_Float16*)alloc((size_t)BV * N_ * SF * 2);
  _Float16* attn16 = (_Float16*)alloc((size_t)BV * 16 * N_ * 2);
  float*    etmp   = (float*)   alloc((size_t)BV * S_ * N_ * 4);
  _Float16* q16    = (_Float16*)alloc((size_t)BV * 16 * Q_ * 2);
  float*    fullb  = (float*)   alloc((size_t)BV * S_ * SF * 4);
  float*    fulln  = (float*)   alloc((size_t)BV * S_ * SF * 4);
  float*    upd    = (float*)   alloc((size_t)BV * 16 * SF * 4);
  float*    sview  = (float*)   alloc((size_t)BV * DV * 4);
  float*    sattr  = (float*)   alloc((size_t)B_ * S_ * DA * 4);

  // attn^T padding rows 7..15 must stay zero (rows 0..6 overwritten each step)
  hipMemsetAsync(attn16, 0, (size_t)BV * 16 * N_ * 2, stream);
  wconv_kernel<<<(F_ * (Q_ + SF) + 255) / 256, 256, 0, stream>>>(W_key, W_val, WkT, WvT);
  init_slots_kernel<<<(BV * DV + B_ * S_ * DA + 255) / 256, 256, 0, stream>>>(
      nview, nobj, nbck, vloc, vls, aol, aos, abl, absc, sview, sattr);
  ln_proj_kernel<<<BV * (N_ / 64), 256, 0, stream>>>(x, ln_in_g, ln_in_b, WkT, WvT, key16, valT);

  for (int step = 0; step < NSTEP; ++step) {
    slots_pre_kernel<<<BV, 256, 0, stream>>>(sview, sattr, ln_qry_g, ln_qry_b, W_qry, fullb, q16);
    attn_kernel<<<BV, 256, 0, stream>>>(key16, q16, attn16, etmp);
    updates_kernel<<<BV, 384, 0, stream>>>(attn16, valT, upd);
    gru_mlp_kernel<<<BV, 256, 0, stream>>>(upd, fullb, gWi, gWh, gbi, gbh,
                                           ln_res_g, ln_res_b, W1, b1, W2, b2, fulln);
    means_kernel<<<(BV * DV + B_ * S_ * DA + 255) / 256, 256, 0, stream>>>(fulln, sview, sattr, out);
  }
}
